// _DeformConv2d_51960514347510
// MI455X (gfx1250) — matrix operand order repaired and checked by reading, not compiled, not run on hardware
//
#include <hip/hip_runtime.h>
#include <hip/hip_bf16.h>

// ---------------------------------------------------------------------------
// DeformConv2d for MI455X (gfx1250, wave32, WMMA).
//   B=4, C=64, H=W=128, OUTC=64, KS=3 -> main conv is a (65536 x 576)@(576 x 64)
//   GEMM with K = 576 = 18*32  => 18 x V_WMMA_F32_16X16X32_BF16 per 16x16 tile.
// ---------------------------------------------------------------------------

typedef __attribute__((ext_vector_type(16))) __bf16 v16bf;
typedef __attribute__((ext_vector_type(8)))  float  v8f;

#define B_     4
#define C_     64
#define H_     128
#define W_     128
#define OUTC_  64
#define KTOT   576            // C_ * 9
#define KSTEPS 18             // KTOT / 32 (exact)
#define NPOS   (B_ * H_ * W_) // 65536

__device__ __forceinline__ unsigned short f2bf(float f) {
    unsigned int u = __float_as_uint(f);
    u += 0x7FFFu + ((u >> 16) & 1u);      // round-to-nearest-even
    return (unsigned short)(u >> 16);
}
__device__ __forceinline__ int iclamp(int v, int lo, int hi) {
    return v < lo ? lo : (v > hi ? hi : v);
}

// ---------------------------------------------------------------------------
// Kernel 1: pre-swizzle conv_w (fp32 OIHW) into bf16 WMMA B-fragment layout.
//   Wt[((ocg*18 + kk)*32 + lane)*16 + e],  lane: g=lane/16, N=lane%16,
//   B element [K = kk*32 + g*16 + e][N]  (row-of-B striped across lanes).
// ---------------------------------------------------------------------------
__global__ void wprep_kernel(const float* __restrict__ conv_w,
                             unsigned short* __restrict__ Wt) {
    int idx = blockIdx.x * blockDim.x + threadIdx.x;
    if (idx >= 4 * KSTEPS * 512) return;
    int e    = idx & 15;
    int lane = (idx >> 4) & 31;
    int kk   = (idx >> 9) % KSTEPS;
    int ocg  = (idx >> 9) / KSTEPS;
    int K  = ((lane >> 4) << 4) + e;
    int k  = kk * 32 + K;                 // k = c*9 + n, max 575 < 576
    int c  = k / 9, n = k % 9;
    int oc = (ocg << 4) + (lane & 15);
    Wt[idx] = f2bf(conv_w[(oc * C_ + c) * 9 + n]);
}

// ---------------------------------------------------------------------------
// Kernel 2: offset conv (64 -> 18 ch, 3x3, pad 1). One thread per position;
// p_w accesses are block-uniform -> scalar loads. Output position-major.
// ---------------------------------------------------------------------------
__global__ void offset_kernel(const float* __restrict__ x,
                              const float* __restrict__ p_w,
                              const float* __restrict__ p_b,
                              float* __restrict__ offs) {
    int p = blockIdx.x * blockDim.x + threadIdx.x;
    if (p >= NPOS) return;
    int b = p >> 14;
    int h = (p >> 7) & 127;
    int w = p & 127;
    float acc[18];
#pragma unroll
    for (int ch = 0; ch < 18; ++ch) acc[ch] = p_b[ch];
    const float* xb = x + (size_t)b * C_ * H_ * W_;
    for (int c = 0; c < C_; ++c) {
        const float* xc = xb + c * (H_ * W_);
#pragma unroll
        for (int ki = 0; ki < 3; ++ki) {
            int iy = h - 1 + ki;
            if ((unsigned)iy >= (unsigned)H_) continue;
#pragma unroll
            for (int kj = 0; kj < 3; ++kj) {
                int ix = w - 1 + kj;
                if ((unsigned)ix >= (unsigned)W_) continue;
                float v = xc[iy * W_ + ix];
#pragma unroll
                for (int ch = 0; ch < 18; ++ch)
                    acc[ch] = fmaf(v, p_w[(ch * C_ + c) * 9 + ki * 3 + kj], acc[ch]);
            }
        }
    }
#pragma unroll
    for (int ch = 0; ch < 18; ++ch) offs[p * 18 + ch] = acc[ch];
}

// ---------------------------------------------------------------------------
// Kernel 3: fused bilinear sampling + WMMA GEMM.
// Workgroup = 128 threads (4 waves) owns 16 consecutive positions (M-tile).
// Wave w handles output-channel group w (N-tile of 16). 18 bf16 WMMAs/wave.
// ---------------------------------------------------------------------------
__global__ void __launch_bounds__(128)
deform_wmma_kernel(const float* __restrict__ x,
                   const float* __restrict__ offs,
                   const unsigned short* __restrict__ Wt,
                   const float* __restrict__ conv_b,
                   float* __restrict__ out) {
    // A fragments: [kk][lane][e] bf16 -> per-lane contiguous 32B (2x ds_load_b128)
    __shared__ __align__(64) unsigned short Amat[KSTEPS * 512];
    __shared__ int   sX0[144], sX1[144], sY0[144], sY1[144];
    __shared__ float sWlt[144], sWrb[144], sWlb[144], sWrt[144];

    const int tid   = threadIdx.x;
    const int pbase = blockIdx.x * 16;          // 16 consecutive w in one (b,h) row
    const int b  = pbase >> 14;
    const int h  = (pbase >> 7) & 127;
    const int w0 = pbase & 127;

    // ---- Phase a: bilinear corners + weights for 16 positions x 9 taps ----
    for (int e = tid; e < 144; e += 128) {
        int m = e / 9, n = e % 9;
        int p = pbase + m;
        float px = offs[p * 18 + n]     + (float)(h + 1)      + (float)(n / 3 - 1);
        float py = offs[p * 18 + 9 + n] + (float)(w0 + m + 1) + (float)(n % 3 - 1);
        float fx = floorf(px), fy = floorf(py);
        int x0 = iclamp((int)fx,     0, 129);
        int x1 = iclamp((int)fx + 1, 0, 129);
        int y0 = iclamp((int)fy,     0, 129);
        int y1 = iclamp((int)fy + 1, 0, 129);
        float pxc = fminf(fmaxf(px, 0.f), 129.f);
        float pyc = fminf(fmaxf(py, 0.f), 129.f);
        float ax = 1.f + ((float)x0 - pxc);
        float bx = 1.f - ((float)x1 - pxc);
        float ay = 1.f + ((float)y0 - pyc);
        float by = 1.f - ((float)y1 - pyc);
        sX0[e] = x0; sX1[e] = x1; sY0[e] = y0; sY1[e] = y1;
        sWlt[e] = ax * ay; sWrb[e] = bx * by;
        sWlb[e] = ax * by; sWrt[e] = bx * ay;
    }
    __syncthreads();

    // ---- Phase b: gather + bilinear blend, pack straight into A-fragment ----
    const float* xb = x + (size_t)b * C_ * H_ * W_;
    for (int idx = tid; idx < 16 * KTOT; idx += 128) {
        int m = idx / KTOT;
        int k = idx % KTOT;                   // k = c*9 + n
        int c = k / 9, n = k % 9;
        int e9 = m * 9 + n;
        const float* xc = xb + c * (H_ * W_);
        int x0 = sX0[e9], x1 = sX1[e9], y0 = sY0[e9], y1 = sY1[e9];
        float v = 0.f;
        if (x0 >= 1 && x0 <= H_) {            // padded coord -> real row x0-1
            const float* row = xc + (x0 - 1) * W_;
            if (y0 >= 1 && y0 <= W_) v += sWlt[e9] * row[y0 - 1];
            if (y1 >= 1 && y1 <= W_) v += sWlb[e9] * row[y1 - 1];
        }
        if (x1 >= 1 && x1 <= H_) {
            const float* row = xc + (x1 - 1) * W_;
            if (y1 >= 1 && y1 <= W_) v += sWrb[e9] * row[y1 - 1];
            if (y0 >= 1 && y0 <= W_) v += sWrt[e9] * row[y0 - 1];
        }
        // A layout (contiguous k, matches Wt): lane = (K/16)*16+m, e = K%16
        int K    = k & 31, kk = k >> 5;
        int lane = ((K >> 4) << 4) + m;
        int el   = K & 15;
        Amat[(kk << 9) + (lane << 4) + el] = f2bf(v);
    }
    __syncthreads();

    // ---- Phase c: 18 x v_wmma_f32_16x16x32_bf16 per wave ----
    const int wave = tid >> 5;
    const int lane = tid & 31;
    const v16bf* Ap = (const v16bf*)Amat;
    const v16bf* Bp = (const v16bf*)(Wt + (size_t)wave * KSTEPS * 512);
    v8f acc = {};
#pragma unroll
    for (int kk = 0; kk < KSTEPS; ++kk) {
        v16bf a  = Ap[kk * 32 + lane];
        v16bf bm = Bp[kk * 32 + lane];
        acc = __builtin_amdgcn_wmma_f32_16x16x32_bf16(
                  false, a, false, bm, (short)0, acc, false, false);
    }

    // ---- Phase d: C/D layout (lanes 0-15: M=r, lanes 16-31: M=8+r), + bias ----
    int ocl  = lane & 15;
    int oc   = (wave << 4) + ocl;
    float bias = conv_b[oc];
    int mbase  = (lane >> 4) << 3;
#pragma unroll
    for (int r = 0; r < 8; ++r) {
        int m  = mbase + r;
        int ww = w0 + m;
        out[(((size_t)b * OUTC_ + oc) * H_ + h) * W_ + ww] = acc[r] + bias;
    }
}

// ---------------------------------------------------------------------------
extern "C" void kernel_launch(void* const* d_in, const int* in_sizes, int n_in,
                              void* d_out, int out_size, void* d_ws, size_t ws_size,
                              hipStream_t stream) {
    const float* x      = (const float*)d_in[0];
    const float* p_w    = (const float*)d_in[1];
    const float* p_b    = (const float*)d_in[2];
    const float* conv_w = (const float*)d_in[3];
    const float* conv_b = (const float*)d_in[4];
    float* out = (float*)d_out;

    // Workspace layout: [bf16 B-fragments 73728B][offsets fp32 65536*18]
    unsigned short* Wt = (unsigned short*)d_ws;
    float* offs = (float*)((char*)d_ws + 4 * KSTEPS * 512 * sizeof(unsigned short));

    wprep_kernel<<<(4 * KSTEPS * 512 + 255) / 256, 256, 0, stream>>>(conv_w, Wt);
    offset_kernel<<<NPOS / 128, 128, 0, stream>>>(x, p_w, p_b, offs);
    deform_wmma_kernel<<<NPOS / 16, 128, 0, stream>>>(x, offs, Wt, conv_b, out);
}
